// Gromov_Wasserstein_Loss_62972810494284
// MI455X (gfx1250) — matrix operand order repaired and checked by reading, compiled, not run on hardware
//
#include <hip/hip_runtime.h>
#include <hip/hip_bf16.h>

// ---------------------------------------------------------------------------
// Gromov-Wasserstein loss, N=4096, D=256, on MI455X (gfx1250, wave32, WMMA).
// Dominant cost: 10x 4096^3 GEMMs -> v_wmma_f32_16x16x32_bf16 (f32 accum).
// All big matmuls are NT GEMMs (A,B row-major, C=A*B^T) so every WMMA
// fragment is a contiguous per-lane chunk; 64MiB matrices are L2-resident.
// Wave tile 64x64 (4x4 WMMA tiles) -> ~32 FLOP per byte pulled from L2.
// ---------------------------------------------------------------------------

typedef __attribute__((ext_vector_type(16))) __bf16 v16bf;
typedef __attribute__((ext_vector_type(8)))  float  v8f;

#define NMAT 4096
#define FDIM 256
#define INV_N (1.0f / 4096.0f)
#define INV_EPS 20.0f            // 1/GW_EPSILON, GW_EPSILON = 0.05
#define GW_ITERS 5
#define SINK_ITERS 20
#define LOG_EPS 1e-10f

// ------------------------------- helpers -----------------------------------

static __device__ inline __bf16 f2bf(float f) {
  union { float f; unsigned u; } a; a.f = f;
  unsigned r = (a.u + 0x7FFFu + ((a.u >> 16) & 1u)) >> 16;  // RNE
  union { unsigned short s; __bf16 b; } o; o.s = (unsigned short)r;
  return o.b;
}

static __device__ inline float block_sum256(float val) {
  __shared__ float s[256];
  s[threadIdx.x] = val;
  __syncthreads();
  for (int off = 128; off > 0; off >>= 1) {
    if ((int)threadIdx.x < off) s[threadIdx.x] += s[threadIdx.x + off];
    __syncthreads();
  }
  float r = s[0];
  __syncthreads();
  return r;
}

static __device__ inline float block_min256(float val) {
  __shared__ float s[256];
  s[threadIdx.x] = val;
  __syncthreads();
  for (int off = 128; off > 0; off >>= 1) {
    if ((int)threadIdx.x < off) s[threadIdx.x] = fminf(s[threadIdx.x], s[threadIdx.x + off]);
    __syncthreads();
  }
  float r = s[0];
  __syncthreads();
  return r;
}

// A fragment (16x32 MxK, bf16): lane half h, row l16. Element e holds
// K = k0 + 16h + e  -> 16 contiguous bf16 (matches B fragment order)
static __device__ inline v16bf ld_afrag(const __bf16* p) {
  union { struct { uint4 lo, hi; } s; v16bf v; } u;
  u.s.lo = *reinterpret_cast<const uint4*>(p);
  u.s.hi = *reinterpret_cast<const uint4*>(p + 8);
  return u.v;
}

// B fragment (32x16 KxN, bf16): lane half h, col l16. Element e holds
// K = k0 + 16h + e  -> 16 contiguous bf16 (one row of the NT operand).
static __device__ inline v16bf ld_bfrag(const __bf16* p) {
  union { struct { uint4 lo, hi; } s; v16bf v; } u;
  u.s.lo = *reinterpret_cast<const uint4*>(p);
  u.s.hi = *reinterpret_cast<const uint4*>(p + 8);
  return u.v;
}

// C/D 16x16 f32: lane half h, N = l16, VGPR r -> M = r + 8h.
static __device__ inline void st_ctile(float* __restrict__ C, int m0, int n0,
                                       v8f acc, int half, int l16) {
#pragma unroll
  for (int r = 0; r < 8; ++r)
    C[(size_t)(m0 + 8 * half + r) * NMAT + n0 + l16] = acc[r];
}

// ------------------------------ kernels ------------------------------------

// Row-normalize f32 [4096,256] -> bf16. One block per row.
__global__ void __launch_bounds__(256)
rownorm_k(const float* __restrict__ X, __bf16* __restrict__ Y) {
  int i = blockIdx.x;
  float x = X[(size_t)i * FDIM + threadIdx.x];
  float ss = block_sum256(x * x);
  float inv = rsqrtf(ss);
  Y[(size_t)i * FDIM + threadIdx.x] = f2bf(x * inv);
}

// NT GEMM: C[m][n] = sum_k A[m,k] * B[n,k].  A:MxK, B:NxK row-major bf16,
// C: 4096x4096 f32.  Block = 8 waves (4 in M, 2 in N), block tile 256x128,
// wave tile 64x64 = 4x4 WMMA 16x16x32 bf16 tiles.  K multiple of 32.
__global__ void __launch_bounds__(256)
gemm_nt_k(const __bf16* __restrict__ A, const __bf16* __restrict__ B,
          float* __restrict__ C, int K) {
  const int tid  = threadIdx.x;
  const int wave = tid >> 5;
  const int lane = tid & 31;
  const int half = lane >> 4;
  const int l16  = lane & 15;
  const int wm   = wave >> 1;            // 0..3
  const int wn   = wave & 1;             // 0..1
  const int m0   = blockIdx.y * 256 + wm * 64;
  const int n0   = blockIdx.x * 128 + wn * 64;

  const __bf16* pa[4];
  const __bf16* pb[4];
#pragma unroll
  for (int t = 0; t < 4; ++t) {
    pa[t] = A + (size_t)(m0 + 16 * t + l16) * K + 16 * half;
    pb[t] = B + (size_t)(n0 + 16 * t + l16) * K + 16 * half;
  }

  v8f acc[4][4];
#pragma unroll
  for (int i = 0; i < 4; ++i)
#pragma unroll
    for (int j = 0; j < 4; ++j)
      acc[i][j] = (v8f){0.f, 0.f, 0.f, 0.f, 0.f, 0.f, 0.f, 0.f};

  for (int k0 = 0; k0 < K; k0 += 32) {
    v16bf af[4], bfr[4];
#pragma unroll
    for (int t = 0; t < 4; ++t) {
      af[t]  = ld_afrag(pa[t] + k0);
      bfr[t] = ld_bfrag(pb[t] + k0);
    }
    __builtin_prefetch(pa[0] + k0 + 64, 0, 3);   // global_prefetch_b8 (near)
    __builtin_prefetch(pb[0] + k0 + 64, 0, 3);
#pragma unroll
    for (int i = 0; i < 4; ++i)
#pragma unroll
      for (int j = 0; j < 4; ++j)
        acc[i][j] = __builtin_amdgcn_wmma_f32_16x16x32_bf16(
            false, af[i], false, bfr[j], (short)0, acc[i][j], false, false);
  }

#pragma unroll
  for (int i = 0; i < 4; ++i)
#pragma unroll
    for (int j = 0; j < 4; ++j)
      st_ctile(C, m0 + 16 * i, n0 + 16 * j, acc[i][j], half, l16);
}

// Grid-stride min over Gram (max(1-G) = 1 - min(G)). 1024 partials.
__global__ void __launch_bounds__(256)
reduce_min_k(const float* __restrict__ X, float* __restrict__ partial) {
  const size_t count = (size_t)NMAT * NMAT;
  float m = 3.4e38f;
  for (size_t i = (size_t)blockIdx.x * blockDim.x + threadIdx.x; i < count;
       i += (size_t)gridDim.x * blockDim.x)
    m = fminf(m, X[i]);
  float bm = block_min256(m);
  if (threadIdx.x == 0) partial[blockIdx.x] = bm;
}

__global__ void __launch_bounds__(256)
finish_max_k(const float* __restrict__ partial, float* __restrict__ outMax) {
  float m = fminf(fminf(partial[threadIdx.x],       partial[threadIdx.x + 256]),
                  fminf(partial[threadIdx.x + 512], partial[threadIdx.x + 768]));
  float bm = block_min256(m);
  if (threadIdx.x == 0) outMax[0] = 1.0f - bm;   // max of daa = 1 - min(G)
}

// Build bf16 hC = hScale*(1-G)/max and constC vector (1/n)*sum_j c^2 per row.
__global__ void __launch_bounds__(256)
make_hc_k(const float* __restrict__ G, const float* __restrict__ maxPtr,
          __bf16* __restrict__ H, float* __restrict__ vec, float hScale) {
  int i = blockIdx.x;
  float inv = 1.0f / maxPtr[0];
  float acc = 0.f;
  for (int j = threadIdx.x; j < NMAT; j += 256) {
    float c = (1.0f - G[(size_t)i * NMAT + j]) * inv;
    H[(size_t)i * NMAT + j] = f2bf(hScale * c);
    acc += c * c;
  }
  float s = block_sum256(acc);
  if (threadIdx.x == 0) vec[i] = s * INV_N;
}

__global__ void __launch_bounds__(256)
fill_t0_k(float* __restrict__ T, __bf16* __restrict__ Tb) {
  size_t idx = (size_t)blockIdx.x * 256 + threadIdx.x;
  const float t0 = INV_N * INV_N;
  T[idx]  = t0;
  Tb[idx] = f2bf(t0);
}

__global__ void __launch_bounds__(256)
f32_to_bf16_k(const float* __restrict__ S, __bf16* __restrict__ D) {
  size_t idx = (size_t)blockIdx.x * 256 + threadIdx.x;
  D[idx] = f2bf(S[idx]);
}

// In place: K = exp(-(vecA[i]+vecB[j]-S)/eps)
__global__ void __launch_bounds__(256)
make_K_k(float* __restrict__ S, const float* __restrict__ vecA,
         const float* __restrict__ vecB) {
  size_t idx = (size_t)blockIdx.x * 256 + threadIdx.x;
  int i = (int)(idx >> 12);
  int j = (int)(idx & (NMAT - 1));
  float t = vecA[i] + vecB[j] - S[idx];
  S[idx] = __expf(-t * INV_EPS);
}

__global__ void __launch_bounds__(256)
fill_ones_k(float* __restrict__ p) {
  p[(size_t)blockIdx.x * 256 + threadIdx.x] = 1.0f;
}

// u[i] = (1/n) / sum_j K[i][j] v[j]   (one block per row)
__global__ void __launch_bounds__(256)
kv_u_k(const float* __restrict__ Km, const float* __restrict__ v,
       float* __restrict__ u) {
  int i = blockIdx.x;
  float acc = 0.f;
  const float* row = Km + (size_t)i * NMAT;
  for (int j = threadIdx.x; j < NMAT; j += 256) acc += row[j] * v[j];
  float s = block_sum256(acc);
  if (threadIdx.x == 0) u[i] = INV_N / s;
}

// Partial column sums of K^T u: part[chunk][j] over 512-row chunks (coalesced)
__global__ void __launch_bounds__(256)
ktu_part_k(const float* __restrict__ Km, const float* __restrict__ u,
           float* __restrict__ part) {
  int j  = blockIdx.x * 256 + threadIdx.x;
  int i0 = blockIdx.y * 512;
  float acc = 0.f;
#pragma unroll 4
  for (int i = i0; i < i0 + 512; ++i) acc += Km[(size_t)i * NMAT + j] * u[i];
  part[(size_t)blockIdx.y * NMAT + j] = acc;
}

__global__ void __launch_bounds__(256)
ktu_fin_k(const float* __restrict__ part, float* __restrict__ v) {
  int j = blockIdx.x * 256 + threadIdx.x;
  float acc = 0.f;
#pragma unroll
  for (int c = 0; c < 8; ++c) acc += part[(size_t)c * NMAT + j];
  v[j] = INV_N / acc;
}

// T = u (outer) K (outer) v ; also bf16 copy for the next NT GEMM.
__global__ void __launch_bounds__(256)
make_T_k(const float* __restrict__ Km, const float* __restrict__ u,
         const float* __restrict__ v, float* __restrict__ T,
         __bf16* __restrict__ Tb) {
  size_t idx = (size_t)blockIdx.x * 256 + threadIdx.x;
  int i = (int)(idx >> 12);
  int j = (int)(idx & (NMAT - 1));
  float t = u[i] * Km[idx] * v[j];
  T[idx]  = t;
  Tb[idx] = f2bf(t);
}

__global__ void __launch_bounds__(256)
loss_k(const float* __restrict__ T, float* __restrict__ out) {
  float acc = 0.f;
  for (int i = threadIdx.x; i < NMAT; i += 256)
    acc += __logf(T[(size_t)i * (NMAT + 1)] + LOG_EPS);
  float s = block_sum256(acc);
  if (threadIdx.x == 0) out[0] = -s;   // GAMMA = 1
}

// ------------------------------ launcher -----------------------------------

extern "C" void kernel_launch(void* const* d_in, const int* in_sizes, int n_in,
                              void* d_out, int out_size, void* d_ws, size_t ws_size,
                              hipStream_t stream) {
  (void)in_sizes; (void)n_in; (void)out_size; (void)ws_size;
  const float* f1 = (const float*)d_in[0];
  const float* f2 = (const float*)d_in[1];
  float* out = (float*)d_out;

  // Workspace layout (needs ~262 MiB; matrices 64 MiB f32 / 32 MiB bf16):
  char* w = (char*)d_ws;
  float*  bufF = (float*)(w);                       // Gram / Wt / S / K
  float*  bufT = (float*)(w + (64ull  << 20));      // transport plan T
  __bf16* hC1b = (__bf16*)(w + (128ull << 20));
  __bf16* hC2b = (__bf16*)(w + (160ull << 20));     // 2*dbb in bf16
  __bf16* Tb   = (__bf16*)(w + (192ull << 20));
  __bf16* Wtb  = (__bf16*)(w + (224ull << 20));
  __bf16* X1b  = (__bf16*)(w + (256ull << 20));     // 2 MiB each
  __bf16* X2b  = (__bf16*)(w + (258ull << 20));
  float*  vecA = (float*)(w + (260ull << 20));      // small vectors
  float*  vecB = vecA + NMAT;
  float*  uvec = vecB + NMAT;
  float*  vvec = uvec + NMAT;
  float*  part = vvec + NMAT;                       // 1024 min partials
  float*  maxA = part + 1024;
  float*  maxB = maxA + 1;
  float*  ktuP = (float*)(w + (261ull << 20));      // 8*4096 col partials

  const size_t NN = (size_t)NMAT * NMAT;
  const dim3 gemmGrid(NMAT / 128, NMAT / 256);      // 32 x 16 blocks
  const int ewBlocks = (int)(NN / 256);             // 65536

  // 1) normalize features -> bf16
  rownorm_k<<<NMAT, 256, 0, stream>>>(f1, X1b);
  rownorm_k<<<NMAT, 256, 0, stream>>>(f2, X2b);

  // 2) Gram 1 -> hC1 (scale 1) + constC vector A
  gemm_nt_k<<<gemmGrid, 256, 0, stream>>>(X1b, X1b, bufF, FDIM);
  reduce_min_k<<<1024, 256, 0, stream>>>(bufF, part);
  finish_max_k<<<1, 256, 0, stream>>>(part, maxA);
  make_hc_k<<<NMAT, 256, 0, stream>>>(bufF, maxA, hC1b, vecA, 1.0f);

  // 3) Gram 2 -> hC2 = 2*dbb + constC vector B
  gemm_nt_k<<<gemmGrid, 256, 0, stream>>>(X2b, X2b, bufF, FDIM);
  reduce_min_k<<<1024, 256, 0, stream>>>(bufF, part);
  finish_max_k<<<1, 256, 0, stream>>>(part, maxB);
  make_hc_k<<<NMAT, 256, 0, stream>>>(bufF, maxB, hC2b, vecB, 2.0f);

  // 4) T0 = p q^T = 1/n^2
  fill_t0_k<<<ewBlocks, 256, 0, stream>>>(bufT, Tb);

  // 5) GW fixed-point iterations
  for (int gw = 0; gw < GW_ITERS; ++gw) {
    // Wt = hC2 * T^T   (NT GEMM; Wt[j][i] = (T*hC2^T)[i][j])
    gemm_nt_k<<<gemmGrid, 256, 0, stream>>>(hC2b, Tb, bufF, NMAT);
    f32_to_bf16_k<<<ewBlocks, 256, 0, stream>>>(bufF, Wtb);
    // S = hC1 * Wt^T = hC1 * (T * hC2^T)
    gemm_nt_k<<<gemmGrid, 256, 0, stream>>>(hC1b, Wtb, bufF, NMAT);
    // K = exp(-(constC - S)/eps)  in place
    make_K_k<<<ewBlocks, 256, 0, stream>>>(bufF, vecA, vecB);
    // Sinkhorn
    fill_ones_k<<<NMAT / 256, 256, 0, stream>>>(uvec);
    fill_ones_k<<<NMAT / 256, 256, 0, stream>>>(vvec);
    for (int s = 0; s < SINK_ITERS; ++s) {
      kv_u_k<<<NMAT, 256, 0, stream>>>(bufF, vvec, uvec);
      ktu_part_k<<<dim3(NMAT / 256, 8), 256, 0, stream>>>(bufF, uvec, ktuP);
      ktu_fin_k<<<NMAT / 256, 256, 0, stream>>>(ktuP, vvec);
    }
    // T = diag(u) K diag(v)
    make_T_k<<<ewBlocks, 256, 0, stream>>>(bufF, uvec, vvec, bufT, Tb);
  }

  // 6) loss = -sum(log(diag(T) + eps))
  loss_k<<<1, 256, 0, stream>>>(bufT, out);
}
